// PointSIFT_res_module_26972394619821
// MI455X (gfx1250) — compile-verified, matrix operand order checked by reading, not run on hardware
//
#include <hip/hip_runtime.h>
#include <hip/hip_bf16.h>

typedef __attribute__((ext_vector_type(16))) _Float16 v16h;
typedef __attribute__((ext_vector_type(8)))  _Float16 v8h;
typedef __attribute__((ext_vector_type(8)))  float    v8f;

#define BT   16
#define NPTS 1024
#define OCH  128
#define ECH  64
#define RAD2 0.04f

// Async global -> LDS copy (CDNA5). ldsoff: byte offset in group segment (VGPR),
// gaddr: 64-bit global byte address (VGPR pair). Tracked by ASYNCcnt.
#define ASYNC_LD_B128(ldsoff, gaddr)                                              \
    asm volatile("global_load_async_to_lds_b128 %0, %1, off"                      \
                 :: "v"(ldsoff), "v"(gaddr) : "memory")
#define WAIT_ASYNC0() asm volatile("s_wait_asynccnt 0x0" ::: "memory")

// ---------------------------------------------------------------------------
// 1) Octant-nearest-neighbor selection: idx[BT][N][8]
// ---------------------------------------------------------------------------
__global__ __launch_bounds__(256)
void select_octant_kernel(const float* __restrict__ xyz, int* __restrict__ idx)
{
    const int bt = blockIdx.y;
    const int n  = blockIdx.x * 256 + threadIdx.x;
    const float* xb = xyz + (size_t)bt * NPTS * 3;
    const float px = xb[n * 3 + 0], py = xb[n * 3 + 1], pz = xb[n * 3 + 2];

    float bd[8];
    int   bi[8];
#pragma unroll
    for (int k = 0; k < 8; ++k) { bd[k] = RAD2; bi[k] = n; }

    __shared__ float sx[256 * 3];
    for (int j0 = 0; j0 < NPTS; j0 += 256) {
        __syncthreads();
        const int t = threadIdx.x;
        sx[t * 3 + 0] = xb[(j0 + t) * 3 + 0];
        sx[t * 3 + 1] = xb[(j0 + t) * 3 + 1];
        sx[t * 3 + 2] = xb[(j0 + t) * 3 + 2];
        __syncthreads();
        for (int jj = 0; jj < 256; ++jj) {
            const float dx = sx[jj * 3 + 0] - px;
            const float dy = sx[jj * 3 + 1] - py;
            const float dz = sx[jj * 3 + 2] - pz;
            const float d  = dx * dx + dy * dy + dz * dz;
            // truncation toward zero matches the reference int cast; dx+1 in (0,2)
            const int oct = ((int)(dx + 1.0f)) * 4 + ((int)(dy + 1.0f)) * 2 + (int)(dz + 1.0f);
            const bool ok = (d > 1e-10f);
            const int  j  = j0 + jj;
#pragma unroll
            for (int k = 0; k < 8; ++k) {
                const bool take = ok && (oct == k) && (d < bd[k]);
                bd[k] = take ? d : bd[k];
                bi[k] = take ? j : bi[k];
            }
        }
    }
    int* op = idx + ((size_t)bt * NPTS + n) * 8;
#pragma unroll
    for (int k = 0; k < 8; ++k) op[k] = bi[k];
}

// ---------------------------------------------------------------------------
// 2) Weight packing: w[o][c][k] (fp32) -> Wp[o][k*C + c] (fp16), K padded to Kp
// ---------------------------------------------------------------------------
__global__ __launch_bounds__(256)
void pack_w_kernel(const float* __restrict__ src, _Float16* __restrict__ dst,
                   int C, int Kp)
{
    const int tid = blockIdx.x * 256 + threadIdx.x;
    if (tid >= 128 * Kp) return;
    const int o  = tid / Kp;
    const int kk = tid % Kp;
    _Float16 v = (_Float16)0.0f;
    if (kk < 2 * C) {
        const int k = kk / C;
        const int c = kk % C;
        v = (_Float16)src[((size_t)o * C + c) * 2 + k];
    }
    dst[(size_t)o * Kp + kk] = v;
}

// ---------------------------------------------------------------------------
// 3) Stage-1 GEMM input: X1[pos=(bt,n,p)][k=(j&1)*67 + c], Kp=160
// ---------------------------------------------------------------------------
__global__ __launch_bounds__(256)
void build_x1_kernel(const float* __restrict__ xyz, const float* __restrict__ pts,
                     const int* __restrict__ idx, _Float16* __restrict__ X1)
{
    const int gid = blockIdx.x * 256 + threadIdx.x;   // over BT*N*8
    const int j   = gid & 7;
    const int n   = (gid >> 3) & (NPTS - 1);
    const int bt  = gid >> 13;
    const int gi  = idx[gid];
    const float* xb = xyz + ((size_t)bt * NPTS + n)  * 3;
    const float* xg = xyz + ((size_t)bt * NPTS + gi) * 3;
    const size_t row = (size_t)(gid >> 1);            // (bt*N+n)*4 + j/2
    _Float16* dst = X1 + row * 160 + (j & 1) * 67;
    dst[0] = (_Float16)(xg[0] - xb[0]);
    dst[1] = (_Float16)(xg[1] - xb[1]);
    dst[2] = (_Float16)(xg[2] - xb[2]);
    const float* pg = pts + ((size_t)bt * NPTS + gi) * ECH;
#pragma unroll 4
    for (int c = 0; c < ECH; ++c) dst[3 + c] = (_Float16)pg[c];
    if (!(j & 1)) {                                   // zero K padding 134..159
        _Float16* pad = X1 + row * 160 + 134;
#pragma unroll
        for (int i = 0; i < 26; ++i) pad[i] = (_Float16)0.0f;
    }
}

// ---------------------------------------------------------------------------
// 4) Stage-2 GEMM input: X4[pos][k=(j&1)*131 + c], Kp=288
// ---------------------------------------------------------------------------
__global__ __launch_bounds__(256)
void build_x4_kernel(const float* __restrict__ xyz, const _Float16* __restrict__ new1,
                     const int* __restrict__ idx, _Float16* __restrict__ X4)
{
    const int gid = blockIdx.x * 256 + threadIdx.x;   // over BT*N*8
    const int j   = gid & 7;
    const int n   = (gid >> 3) & (NPTS - 1);
    const int bt  = gid >> 13;
    const int gi  = idx[gid];
    const float* xb = xyz + ((size_t)bt * NPTS + n)  * 3;
    const float* xg = xyz + ((size_t)bt * NPTS + gi) * 3;
    const size_t row = (size_t)(gid >> 1);
    _Float16* dst = X4 + row * 288 + (j & 1) * 131;
    dst[0] = (_Float16)(xg[0] - xb[0]);
    dst[1] = (_Float16)(xg[1] - xb[1]);
    dst[2] = (_Float16)(xg[2] - xb[2]);
    const _Float16* ng = new1 + ((size_t)bt * NPTS + gi) * OCH;
#pragma unroll 4
    for (int c = 0; c < OCH; ++c) dst[3 + c] = ng[c];
    if (!(j & 1)) {                                   // zero K padding 262..287
        _Float16* pad = X4 + row * 288 + 262;
#pragma unroll
        for (int i = 0; i < 26; ++i) pad[i] = (_Float16)0.0f;
    }
}

// ---------------------------------------------------------------------------
// 5) WMMA GEMM: Y[128 x Npos] = W[128 x Nk] * X[Npos x Nk]^T  (+bias)
//    256 threads (8 wave32) -> 128x128 tile; K chunked by 32.
//    Staging uses CDNA5 global_load_async_to_lds_b128 (ASYNCcnt).
//    MODE 0: repack pairs for next stride-2 conv (fp16, stride knext)
//    MODE 1: store new1 fp16 [pos][128]
//    MODE 2: final ReLU fp32 store into [pos][192] channel-interleaved output
// ---------------------------------------------------------------------------
template <int MODE>
__global__ __launch_bounds__(256)
void wmma_gemm_kernel(const _Float16* __restrict__ Aw, const float* __restrict__ bias,
                      const _Float16* __restrict__ Xin, int Nk,
                      _Float16* __restrict__ OutH, float* __restrict__ OutF, int knext)
{
    // Single shared block so inline-asm LDS byte offsets are well defined:
    //   sA = bytes [0, 8192), sX = bytes [8192, 16384)
    __shared__ __attribute__((aligned(128))) _Float16 smem[8192];
    _Float16* sA = smem;
    _Float16* sX = smem + 4096;

    const int tid  = threadIdx.x;
    const int lane = tid & 31;
    const int wave = tid >> 5;
    const size_t posBase = (size_t)blockIdx.x * 128;

    v8f acc[8];
#pragma unroll
    for (int t = 0; t < 8; ++t)
#pragma unroll
        for (int r = 0; r < 8; ++r) acc[t][r] = 0.0f;

    const int lrow = lane & 15;
    const int kb   = lane & 16;          // K half selected by lane group (0 or 16)

    // Per-thread staging map: rows (rA, rA+64), 16-byte segment seg of the
    // 64-byte (32-half) K-chunk row. 256 threads x 4 transfers = 16 KB/chunk.
    const int rA  = tid >> 2;
    const int seg = tid & 3;
    const unsigned ldsA0 = (unsigned)((rA * 32 + seg * 8) * 2);
    const unsigned ldsA1 = ldsA0 + 64 * 32 * 2;
    const unsigned ldsX0 = ldsA0 + 8192;
    const unsigned ldsX1 = ldsA1 + 8192;

    for (int k0 = 0; k0 < Nk; k0 += 32) {
        __syncthreads();                 // previous chunk fully consumed
        const unsigned long long gA0 =
            (unsigned long long)(Aw + (size_t)rA * Nk + k0 + seg * 8);
        const unsigned long long gA1 =
            (unsigned long long)(Aw + (size_t)(rA + 64) * Nk + k0 + seg * 8);
        const unsigned long long gX0 =
            (unsigned long long)(Xin + (posBase + rA) * (size_t)Nk + k0 + seg * 8);
        const unsigned long long gX1 =
            (unsigned long long)(Xin + (posBase + rA + 64) * (size_t)Nk + k0 + seg * 8);
        ASYNC_LD_B128(ldsA0, gA0);
        ASYNC_LD_B128(ldsA1, gA1);
        ASYNC_LD_B128(ldsX0, gX0);
        ASYNC_LD_B128(ldsX1, gX1);
        WAIT_ASYNC0();
        __syncthreads();

        // A fragment: 16x32 fp16, lane {M=lane&15, Kbase=16*(lane>=16)}, contig halves
        const v16h afrag = *(const v16h*)(&sA[(wave * 16 + lrow) * 32 + kb]);
        // Software-pipeline B fragments one tile ahead of the WMMA consuming them.
        v16h bcur = *(const v16h*)(&sX[lrow * 32 + kb]);
#pragma unroll
        for (int t = 0; t < 8; ++t) {
            v16h bnext = bcur;
            if (t < 7) bnext = *(const v16h*)(&sX[((t + 1) * 16 + lrow) * 32 + kb]);
            acc[t] = __builtin_amdgcn_wmma_f32_16x16x32_f16(
                false, afrag, false, bcur, (short)0, acc[t], false, false);
            bcur = bnext;
        }
    }

    // Epilogue. D layout: lanes 0-15: VGPR r -> M=r; lanes 16-31: VGPR r -> M=8+r.
    // Per lane the 8 rows are 8 consecutive output channels -> packed stores.
    const int ncol = lane & 15;
    const int ci0  = wave * 16 + (lane >> 4) * 8;
    float bv[8];
#pragma unroll
    for (int r = 0; r < 8; ++r) bv[r] = bias[ci0 + r];

#pragma unroll
    for (int t = 0; t < 8; ++t) {
        const size_t q = posBase + t * 16 + ncol;     // global position index
        if (MODE == 2) {
            float* d = OutF + q * 192 + ci0;
            float4 f0, f1;
            f0.x = fmaxf(acc[t][0] + bv[0], 0.0f);
            f0.y = fmaxf(acc[t][1] + bv[1], 0.0f);
            f0.z = fmaxf(acc[t][2] + bv[2], 0.0f);
            f0.w = fmaxf(acc[t][3] + bv[3], 0.0f);
            f1.x = fmaxf(acc[t][4] + bv[4], 0.0f);
            f1.y = fmaxf(acc[t][5] + bv[5], 0.0f);
            f1.z = fmaxf(acc[t][6] + bv[6], 0.0f);
            f1.w = fmaxf(acc[t][7] + bv[7], 0.0f);
            *(float4*)(d)     = f0;
            *(float4*)(d + 4) = f1;
        } else {
            v8h h;
#pragma unroll
            for (int r = 0; r < 8; ++r) h[r] = (_Float16)(acc[t][r] + bv[r]);
            _Float16* d = (MODE == 0)
                              ? (OutH + (q >> 1) * (size_t)knext + (q & 1) * 128 + ci0)
                              : (OutH + q * 128 + ci0);
            *(v8h*)d = h;                             // one 16B store per tile
        }
    }
}

// ---------------------------------------------------------------------------
// 6) Output tail: copy xyz (tuple output 0) and ReLU(points) -> channels 128..191
// ---------------------------------------------------------------------------
__global__ __launch_bounds__(256)
void copy_xyz_kernel(const float* __restrict__ xyz, float* __restrict__ out)
{
    const int tid = blockIdx.x * 256 + threadIdx.x;
    if (tid < BT * NPTS * 3) out[tid] = xyz[tid];
}

__global__ __launch_bounds__(256)
void relu_pts_kernel(const float* __restrict__ pts, float* __restrict__ outCh)
{
    const int tid = blockIdx.x * 256 + threadIdx.x;   // over BT*N*64
    const int p = tid >> 6;
    const int c = tid & 63;
    outCh[(size_t)p * 192 + 128 + c] = fmaxf(pts[tid], 0.0f);
}

// ---------------------------------------------------------------------------
extern "C" void kernel_launch(void* const* d_in, const int* in_sizes, int n_in,
                              void* d_out, int out_size, void* d_ws, size_t ws_size,
                              hipStream_t stream)
{
    const float* xyz = (const float*)d_in[0];
    const float* pts = (const float*)d_in[1];
    const float* w1a = (const float*)d_in[2];  const float* b1a = (const float*)d_in[3];
    const float* w1b = (const float*)d_in[4];  const float* b1b = (const float*)d_in[5];
    const float* w1c = (const float*)d_in[6];  const float* b1c = (const float*)d_in[7];
    const float* w2a = (const float*)d_in[8];  const float* b2a = (const float*)d_in[9];
    const float* w2b = (const float*)d_in[10]; const float* b2b = (const float*)d_in[11];
    const float* w2c = (const float*)d_in[12]; const float* b2c = (const float*)d_in[13];

    char* ws = (char*)d_ws;
    int*      idx = (int*)(ws + 0);                       // 16*1024*8 ints = 512 KB
    _Float16* W1  = (_Float16*)(ws + 524288);             // 128*160
    _Float16* W2  = W1 + 128 * 160;                       // 128*256
    _Float16* W3  = W2 + 128 * 256;                       // 128*256
    _Float16* W4  = W3 + 128 * 256;                       // 128*288
    _Float16* W5  = W4 + 128 * 288;                       // 128*256
    _Float16* W6  = W5 + 128 * 256;                       // 128*256
    _Float16* X1  = (_Float16*)(ws + 901120);             // 65536x160 / reused 65536x288
    _Float16* X2  = (_Float16*)(ws + 38649856);           // 32768x256 (reused stage 2)
    _Float16* X3  = (_Float16*)(ws + 55427072);           // 16384x256 (reused stage 2)
    _Float16* NW1 = (_Float16*)(ws + 63815680);           // 16384x128 stage-1 result

    float* outF  = (float*)d_out;                         // [BT*N*3] xyz passthrough
    float* outCh = outF + BT * NPTS * 3;                  // [BT*N][192] features

    // Neighbor selection + weight packing (independent, small)
    select_octant_kernel<<<dim3(4, BT), 256, 0, stream>>>(xyz, idx);
    pack_w_kernel<<<(128 * 160 + 255) / 256, 256, 0, stream>>>(w1a, W1,  67, 160);
    pack_w_kernel<<<(128 * 256 + 255) / 256, 256, 0, stream>>>(w1b, W2, 128, 256);
    pack_w_kernel<<<(128 * 256 + 255) / 256, 256, 0, stream>>>(w1c, W3, 128, 256);
    pack_w_kernel<<<(128 * 288 + 255) / 256, 256, 0, stream>>>(w2a, W4, 131, 288);
    pack_w_kernel<<<(128 * 256 + 255) / 256, 256, 0, stream>>>(w2b, W5, 128, 256);
    pack_w_kernel<<<(128 * 256 + 255) / 256, 256, 0, stream>>>(w2c, W6, 128, 256);

    // Stage 1: gather -> conv1a -> conv1b -> conv1c
    build_x1_kernel<<<512, 256, 0, stream>>>(xyz, pts, idx, X1);
    wmma_gemm_kernel<0><<<512, 256, 0, stream>>>(W1, b1a, X1, 160, X2, nullptr, 256);
    wmma_gemm_kernel<0><<<256, 256, 0, stream>>>(W2, b1b, X2, 256, X3, nullptr, 256);
    wmma_gemm_kernel<1><<<128, 256, 0, stream>>>(W3, b1c, X3, 256, NW1, nullptr, 0);

    // Stage 2: re-gather -> conv2a -> conv2b -> conv2c (fused bias+ReLU+store)
    build_x4_kernel<<<512, 256, 0, stream>>>(xyz, NW1, idx, X1);
    wmma_gemm_kernel<0><<<512, 256, 0, stream>>>(W4, b2a, X1, 288, X2, nullptr, 256);
    wmma_gemm_kernel<0><<<256, 256, 0, stream>>>(W5, b2b, X2, 256, X3, nullptr, 256);
    wmma_gemm_kernel<2><<<128, 256, 0, stream>>>(W6, b2c, X3, 256, nullptr, outCh, 0);

    // Tuple output 0 (xyz) + ReLU(points) into channels 128..191
    copy_xyz_kernel<<<(BT * NPTS * 3 + 255) / 256, 256, 0, stream>>>(xyz, outF);
    relu_pts_kernel<<<(BT * NPTS * ECH) / 256, 256, 0, stream>>>(pts, outCh);
}
